// LESModel_52158082842762
// MI455X (gfx1250) — matrix operand order repaired and checked by reading, not compiled, not run on hardware
//
#include <hip/hip_runtime.h>

#define NATOM 2048
#define NRBF  16
#define HID   64
#define CUTOFF 5.0f
// ETA = 0.5*(5.0-0.5)/16 = 0.140625 ; 1/(2*eta^2)
#define INV2ETA2 25.2839506172839506f
#define PI_OVER_CUT 0.62831853071795864f  // pi / CUTOFF

typedef __attribute__((ext_vector_type(16))) _Float16 v16h;
typedef __attribute__((ext_vector_type(8)))  float    v8f;

// ---------------------------------------------------------------- init
__global__ void init_kernel(float* ws) {
    ws[0] = 0.0f;  // sum raw_q
    ws[1] = 0.0f;  // sum E_sr
    ws[2] = 0.0f;  // E_lr
    ws[3] = 0.0f;
}

// ---------------------------------------------------------------- features
// One wave per atom i; lanes stride the j loop; 16 RBF accumulators in VGPRs,
// xor-shuffle reduced across the wave32.
__global__ void feat_kernel(const float* __restrict__ pos, float* __restrict__ feat) {
    __shared__ float spx[NATOM], spy[NATOM], spz[NATOM];
    const int tid = threadIdx.x;
    for (int i = tid; i < NATOM; i += 256) {
        spx[i] = pos[3 * i + 0];
        spy[i] = pos[3 * i + 1];
        spz[i] = pos[3 * i + 2];
    }
    __syncthreads();

    const int wave = tid >> 5;
    const int lane = tid & 31;
    const int atom = blockIdx.x * 8 + wave;

    const float xi = spx[atom], yi = spy[atom], zi = spz[atom];
    float f[NRBF];
#pragma unroll
    for (int r = 0; r < NRBF; ++r) f[r] = 0.0f;

    for (int j = lane; j < NATOM; j += 32) {
        const float dx = spx[j] - xi;
        const float dy = spy[j] - yi;
        const float dz = spz[j] - zi;
        const float d2 = dx * dx + dy * dy + dz * dz;
        if (d2 > 0.0f) {
            const float d = sqrtf(d2);
            if (d < CUTOFF) {
                const float s = 0.5f * (1.0f + __cosf(PI_OVER_CUT * d));
#pragma unroll
                for (int r = 0; r < NRBF; ++r) {
                    const float t = d - (0.5f + 0.3f * (float)r);
                    f[r] += s * __expf(-t * t * INV2ETA2);
                }
            }
        }
    }
#pragma unroll
    for (int r = 0; r < NRBF; ++r) {
        float t = f[r];
        t += __shfl_xor(t, 1);
        t += __shfl_xor(t, 2);
        t += __shfl_xor(t, 4);
        t += __shfl_xor(t, 8);
        t += __shfl_xor(t, 16);
        f[r] = t;
    }
    if (lane == 0) {
#pragma unroll
        for (int r = 0; r < NRBF; ++r) feat[atom * NRBF + r] = f[r];
    }
}

// ---------------------------------------------------------------- WMMA MLP
// One wave handles a 16-atom row tile. Layer1: A = feat 16x16 (K padded to 32),
// Layer2: K=64 in two 32-chunks; HID=64 -> 4 N-tiles of 16. f16 A/B, f32 acc.
// B operands are PRE-PACKED in LDS in per-lane-contiguous WMMA layout:
//   W1 pack: [nt][lane][e]      (zero-padded K>=16)
//   W2 pack: [kc][nt][lane][e]
// so each operand fetch is a single 32-byte v16h load (2x ds_load_b128).
__device__ __forceinline__ float silu(float x) {
    return x / (1.0f + __expf(-x));
}

#define HSTRIDE 68  // sH row stride in floats (16B-aligned rows, bank-skewed)

__device__ __forceinline__ void run_mlp(const _Float16* __restrict__ sW1p,
                                        const _Float16* __restrict__ sW2p,
                                        const float* __restrict__ sB1,
                                        const float* __restrict__ sB2,
                                        const float* __restrict__ sW3, float b3,
                                        const float* __restrict__ feat, int tileBase,
                                        float (*sHw)[HSTRIDE], int lane, float yout[8]) {
    const int laneHi = lane >> 4;
    const int m = lane & 15;
    const int off = laneHi * 8;

    // ---- layer 1 : A = features (16x32, K>=16 zero padded)
    v16h a;
#pragma unroll
    for (int e = 0; e < 8; ++e) a[e] = (_Float16)feat[(tileBase + m) * NRBF + off + e];
#pragma unroll
    for (int e = 8; e < 16; ++e) a[e] = (_Float16)0.0f;

    v8f acc[4];
#pragma unroll
    for (int nt = 0; nt < 4; ++nt) {
        const v16h b = *(const v16h*)(sW1p + ((nt * 32 + lane) << 4));
        v8f c = {};
        acc[nt] = __builtin_amdgcn_wmma_f32_16x16x32_f16(false, a, false, b,
                                                         (short)0, c, false, false);
    }

    // bias + SiLU, stage h1 to LDS (D layout -> row-major f32)
#pragma unroll
    for (int nt = 0; nt < 4; ++nt) {
        const int col = m + 16 * nt;
        const float bb = sB1[col];
#pragma unroll
        for (int v = 0; v < 8; ++v) {
            sHw[v + 8 * laneHi][col] = silu(acc[nt][v] + bb);
        }
    }
    __syncthreads();

    // ---- layer 2 : A from LDS h1 (16x64), two K chunks of 32
    v8f acc2[4];
#pragma unroll
    for (int nt = 0; nt < 4; ++nt) { v8f z = {}; acc2[nt] = z; }

#pragma unroll
    for (int kc = 0; kc < 2; ++kc) {
        // lane reads two contiguous 8-float runs: k = 32kc+off..+7 and 32kc+16+off..+7
        const float* h0 = &sHw[m][32 * kc + off];
        const float* h1 = &sHw[m][32 * kc + 16 + off];
        const float4 p0 = *(const float4*)(h0);
        const float4 p1 = *(const float4*)(h0 + 4);
        const float4 p2 = *(const float4*)(h1);
        const float4 p3 = *(const float4*)(h1 + 4);
        v16h a2;
        a2[0]  = (_Float16)p0.x; a2[1]  = (_Float16)p0.y;
        a2[2]  = (_Float16)p0.z; a2[3]  = (_Float16)p0.w;
        a2[4]  = (_Float16)p1.x; a2[5]  = (_Float16)p1.y;
        a2[6]  = (_Float16)p1.z; a2[7]  = (_Float16)p1.w;
        a2[8]  = (_Float16)p2.x; a2[9]  = (_Float16)p2.y;
        a2[10] = (_Float16)p2.z; a2[11] = (_Float16)p2.w;
        a2[12] = (_Float16)p3.x; a2[13] = (_Float16)p3.y;
        a2[14] = (_Float16)p3.z; a2[15] = (_Float16)p3.w;
#pragma unroll
        for (int nt = 0; nt < 4; ++nt) {
            const v16h b2 = *(const v16h*)(sW2p + (((kc * 4 + nt) * 32 + lane) << 4));
            acc2[nt] = __builtin_amdgcn_wmma_f32_16x16x32_f16(false, a2, false, b2,
                                                              (short)0, acc2[nt],
                                                              false, false);
        }
    }
    __syncthreads();

    // ---- layer 3 : per-lane partial dot with w3, reduce 16 lanes per half
    float p[8];
#pragma unroll
    for (int v = 0; v < 8; ++v) p[v] = 0.0f;
#pragma unroll
    for (int nt = 0; nt < 4; ++nt) {
        const int col = m + 16 * nt;
        const float bb = sB2[col];
        const float w3 = sW3[col];
#pragma unroll
        for (int v = 0; v < 8; ++v) p[v] += silu(acc2[nt][v] + bb) * w3;
    }
#pragma unroll
    for (int v = 0; v < 8; ++v) {
        float t = p[v];
        t += __shfl_xor(t, 1);
        t += __shfl_xor(t, 2);
        t += __shfl_xor(t, 4);
        t += __shfl_xor(t, 8);
        yout[v] = t + b3;  // row = v + 8*laneHi
    }
}

__global__ void mlp_kernel(const float* __restrict__ feat,
                           const float* cw1, const float* cb1, const float* cw2,
                           const float* cb2, const float* cw3, const float* cb3,
                           const float* ew1, const float* eb1, const float* ew2,
                           const float* eb2, const float* ew3, const float* eb3,
                           float* __restrict__ raw_q, float* sum_q, float* sum_esr) {
    __shared__ __align__(32) _Float16 sW1c[4 * 32 * 16];      // packed, K padded
    __shared__ __align__(32) _Float16 sW1e[4 * 32 * 16];
    __shared__ __align__(32) _Float16 sW2c[2 * 4 * 32 * 16];  // packed
    __shared__ __align__(32) _Float16 sW2e[2 * 4 * 32 * 16];
    __shared__ float sB1c[HID], sB2c[HID], sW3c[HID];
    __shared__ float sB1e[HID], sB2e[HID], sW3e[HID];
    __shared__ __align__(16) float sH[4][16][HSTRIDE];

    const int tid = threadIdx.x;
    // pack W1 (B layout, zero-padded K): idx -> (nt, lane, e)
    for (int idx = tid; idx < 4 * 32 * 16; idx += 128) {
        const int e  = idx & 15;
        const int ln = (idx >> 4) & 31;
        const int nt = idx >> 9;
        const int k  = (e & 7) + 8 * (ln >> 4) + 16 * (e >> 3);
        const int n  = (ln & 15) + 16 * nt;
        const bool ok = (k < NRBF);
        sW1c[idx] = ok ? (_Float16)cw1[k * HID + n] : (_Float16)0.0f;
        sW1e[idx] = ok ? (_Float16)ew1[k * HID + n] : (_Float16)0.0f;
    }
    // pack W2 (B layout): idx -> (kc, nt, lane, e)
    for (int idx = tid; idx < 2 * 4 * 32 * 16; idx += 128) {
        const int e  = idx & 15;
        const int ln = (idx >> 4) & 31;
        const int nt = (idx >> 9) & 3;
        const int kc = idx >> 11;
        const int k  = 32 * kc + (e & 7) + 8 * (ln >> 4) + 16 * (e >> 3);
        const int n  = (ln & 15) + 16 * nt;
        sW2c[idx] = (_Float16)cw2[k * HID + n];
        sW2e[idx] = (_Float16)ew2[k * HID + n];
    }
    if (tid < HID) {
        sB1c[tid] = cb1[tid]; sB2c[tid] = cb2[tid]; sW3c[tid] = cw3[tid];
        sB1e[tid] = eb1[tid]; sB2e[tid] = eb2[tid]; sW3e[tid] = ew3[tid];
    }
    __syncthreads();

    const int wave = tid >> 5;
    const int lane = tid & 31;
    const int laneHi = lane >> 4;
    const int tileBase = (blockIdx.x * 4 + wave) * 16;

    float yq[8];
    run_mlp(sW1c, sW2c, sB1c, sB2c, sW3c, cb3[0], feat, tileBase, sH[wave], lane, yq);
    if ((lane & 15) == 0) {
        float local = 0.0f;
#pragma unroll
        for (int v = 0; v < 8; ++v) {
            raw_q[tileBase + v + 8 * laneHi] = yq[v];
            local += yq[v];
        }
        atomicAdd(sum_q, local);
    }
    __syncthreads();

    float ye[8];
    run_mlp(sW1e, sW2e, sB1e, sB2e, sW3e, eb3[0], feat, tileBase, sH[wave], lane, ye);
    if ((lane & 15) == 0) {
        float local = 0.0f;
#pragma unroll
        for (int v = 0; v < 8; ++v) local += ye[v];
        atomicAdd(sum_esr, local);
    }
}

// ---------------------------------------------------------------- Coulomb
__global__ void coul_kernel(const float* __restrict__ pos,
                            const float* __restrict__ raw_q,
                            const float* __restrict__ ws,
                            float* __restrict__ charges_out,
                            float* __restrict__ e_lr_acc) {
    __shared__ float spx[NATOM], spy[NATOM], spz[NATOM], sq[NATOM];
    const int tid = threadIdx.x;
    const float corr = -ws[0] * (1.0f / (float)NATOM);  // TOTAL_CHARGE = 0
    for (int i = tid; i < NATOM; i += 256) {
        spx[i] = pos[3 * i + 0];
        spy[i] = pos[3 * i + 1];
        spz[i] = pos[3 * i + 2];
        const float q = raw_q[i] + corr;
        sq[i] = q;
        if (blockIdx.x == 0) charges_out[i] = q;
    }
    __syncthreads();

    const int wave = tid >> 5;
    const int lane = tid & 31;
    const int atom = blockIdx.x * 8 + wave;
    const float xi = spx[atom], yi = spy[atom], zi = spz[atom];
    const float qi = sq[atom];

    float s = 0.0f;
    for (int j = lane; j < NATOM; j += 32) {
        if (j == atom) continue;
        const float dx = spx[j] - xi;
        const float dy = spy[j] - yi;
        const float dz = spz[j] - zi;
        const float d2 = dx * dx + dy * dy + dz * dz;
        s += sq[j] * __frsqrt_rn(d2);
    }
    s += __shfl_xor(s, 1);
    s += __shfl_xor(s, 2);
    s += __shfl_xor(s, 4);
    s += __shfl_xor(s, 8);
    s += __shfl_xor(s, 16);
    if (lane == 0) atomicAdd(e_lr_acc, 0.5f * qi * s);
}

// ---------------------------------------------------------------- finalize
__global__ void final_kernel(const float* ws, float* out) {
    out[0] = ws[2] + ws[1];  // E_lr + E_sr
}

// ---------------------------------------------------------------- launch
extern "C" void kernel_launch(void* const* d_in, const int* in_sizes, int n_in,
                              void* d_out, int out_size, void* d_ws, size_t ws_size,
                              hipStream_t stream) {
    const float* pos = (const float*)d_in[0];
    const float* cw1 = (const float*)d_in[1];
    const float* cb1 = (const float*)d_in[2];
    const float* cw2 = (const float*)d_in[3];
    const float* cb2 = (const float*)d_in[4];
    const float* cw3 = (const float*)d_in[5];
    const float* cb3 = (const float*)d_in[6];
    const float* ew1 = (const float*)d_in[7];
    const float* eb1 = (const float*)d_in[8];
    const float* ew2 = (const float*)d_in[9];
    const float* eb2 = (const float*)d_in[10];
    const float* ew3 = (const float*)d_in[11];
    const float* eb3 = (const float*)d_in[12];

    float* out  = (float*)d_out;
    float* wsf  = (float*)d_ws;
    float* sum_q   = wsf + 0;
    float* sum_esr = wsf + 1;
    float* e_lr    = wsf + 2;
    float* feat    = wsf + 4;
    float* raw_q   = wsf + 4 + NATOM * NRBF;

    init_kernel<<<1, 1, 0, stream>>>(wsf);
    feat_kernel<<<NATOM / 8, 256, 0, stream>>>(pos, feat);
    mlp_kernel<<<NATOM / 64, 128, 0, stream>>>(feat,
                                               cw1, cb1, cw2, cb2, cw3, cb3,
                                               ew1, eb1, ew2, eb2, ew3, eb3,
                                               raw_q, sum_q, sum_esr);
    coul_kernel<<<NATOM / 8, 256, 0, stream>>>(pos, raw_q, wsf, out + 1, e_lr);
    final_kernel<<<1, 1, 0, stream>>>(wsf, out);
}
